// RNN_76261439308472
// MI455X (gfx1250) — matrix operand order repaired and checked by reading, not compiled, not run on hardware
//
#include <hip/hip_runtime.h>

// ---------------------------------------------------------------------------
// Elman RNN (ReLU) for MI455X / gfx1250, wave32 + v_wmma_f32_16x16x32_bf16.
//   Phase 1: x_proj = X * W_ih^T + (b_ih + b_hh)  -> d_out   (32x64 wave tiles)
//   Phase 2: 512 sequential step kernels: h = relu(xp_t + h * W_hh^T).
//            h_prev (128 KB bf16) streamed into LDS with
//            global_load_async_to_lds_b128 in K-panel order and consumed in a
//            software pipeline: s_wait_asynccnt <= N  +  barrier per panel.
// ---------------------------------------------------------------------------

typedef __attribute__((ext_vector_type(16))) __bf16 v16bf;
typedef __attribute__((ext_vector_type(8)))  __bf16 v8bf;
typedef __attribute__((ext_vector_type(8)))  float  v8f;

#define B_DIM 64
#define S_DIM 512
#define I_DIM 1024
#define H_DIM 1024

// ---- WMMA fragment loaders (contiguous K order k = 16*half + s, same in A and B) -----------------

// A-frag 16x32 bf16, source is fp32 row-major [*, lda]; convert on load.
__device__ __forceinline__ v16bf load_a_from_f32(const float* __restrict__ A,
                                                 int lda, int m0, int k0) {
  const int lane = threadIdx.x & 31;
  const int half = lane >> 4;                 // 0: K offset +0, 1: K offset +16
  const int m    = m0 + (lane & 15);
  const float* __restrict__ base = A + (size_t)m * lda + k0 + half * 16;
  const float4* p0 = reinterpret_cast<const float4*>(base);        // K chunk 0
  const float4* p1 = reinterpret_cast<const float4*>(base + 8);    // K chunk 1
  float4 f0 = p0[0], f1 = p0[1];
  float4 f2 = p1[0], f3 = p1[1];
  v16bf a;
  a[0]  = (__bf16)f0.x; a[1]  = (__bf16)f0.y; a[2]  = (__bf16)f0.z; a[3]  = (__bf16)f0.w;
  a[4]  = (__bf16)f1.x; a[5]  = (__bf16)f1.y; a[6]  = (__bf16)f1.z; a[7]  = (__bf16)f1.w;
  a[8]  = (__bf16)f2.x; a[9]  = (__bf16)f2.y; a[10] = (__bf16)f2.z; a[11] = (__bf16)f2.w;
  a[12] = (__bf16)f3.x; a[13] = (__bf16)f3.y; a[14] = (__bf16)f3.z; a[15] = (__bf16)f3.w;
  return a;
}

// A-frag 16x32 bf16 from bf16 row-major [*, lda] (global or LDS source).
__device__ __forceinline__ v16bf load_a_from_bf16(const __bf16* __restrict__ A,
                                                  int lda, int m0, int k0) {
  const int lane = threadIdx.x & 31;
  const int half = lane >> 4;
  const int m    = m0 + (lane & 15);
  const __bf16* __restrict__ base = A + (size_t)m * lda + k0 + half * 16;
  v8bf lo = *reinterpret_cast<const v8bf*>(base);
  v8bf hi = *reinterpret_cast<const v8bf*>(base + 8);
  v16bf a;
#pragma unroll
  for (int i = 0; i < 8; ++i) { a[i] = lo[i]; a[8 + i] = hi[i]; }
  return a;
}

// B-frag 32x16: B[k][n] = W[n][k] with W row-major [N, ldw] bf16.
// Lane holds 16 contiguous K values for its column -> one aligned 32B load.
__device__ __forceinline__ v16bf load_b_bf16(const __bf16* __restrict__ W,
                                             int ldw, int n0, int k0) {
  const int lane = threadIdx.x & 31;
  const int n    = n0 + (lane & 15);
  const int kb   = k0 + (lane >> 4) * 16;
  return *reinterpret_cast<const v16bf*>(W + (size_t)n * ldw + kb);
}

#define WMMA_BF16(A, Bf, C) \
  __builtin_amdgcn_wmma_f32_16x16x32_bf16(false, (A), false, (Bf), (short)0, (C), false, false)

// ---- Phase 1: x_proj GEMM -> d_out (wave tile 32x64, WG tile 64x256) ------
__global__ __launch_bounds__(256) void xproj_gemm(
    const float*  __restrict__ X,     // [B*S, I] fp32
    const __bf16* __restrict__ Wih,   // [H, I] bf16
    const float*  __restrict__ bias,  // [H] = b_ih + b_hh
    float*        __restrict__ Out)   // [B*S, H] fp32
{
  const int waveId = threadIdx.x >> 5;
  const int m0 = blockIdx.y * 64  + (waveId & 1) * 32;
  const int n0 = blockIdx.x * 256 + (waveId >> 1) * 64;

  v8f acc[2][4];
#pragma unroll
  for (int i = 0; i < 2; ++i)
#pragma unroll
    for (int j = 0; j < 4; ++j) acc[i][j] = v8f{};

  const int lane = threadIdx.x & 31;
  const int mpf  = m0 + (lane & 15);

  for (int k0 = 0; k0 < I_DIM; k0 += 32) {
    // Unconditional speculative prefetch of the streamed A rows, two K-steps
    // ahead (OOB speculative prefetches are silently dropped, ISA 10.5).
    __builtin_prefetch(X + (size_t)mpf * I_DIM + k0 + 64, 0, 3);
    __builtin_prefetch(X + (size_t)(mpf + 16) * I_DIM + k0 + 64, 0, 3);

    v16bf a0 = load_a_from_f32(X, I_DIM, m0,      k0);
    v16bf a1 = load_a_from_f32(X, I_DIM, m0 + 16, k0);
    v16bf b[4];
#pragma unroll
    for (int j = 0; j < 4; ++j) b[j] = load_b_bf16(Wih, I_DIM, n0 + 16 * j, k0);
#pragma unroll
    for (int j = 0; j < 4; ++j) {
      acc[0][j] = WMMA_BF16(a0, b[j], acc[0][j]);
      acc[1][j] = WMMA_BF16(a1, b[j], acc[1][j]);
    }
  }

  const int half = lane >> 4;
#pragma unroll
  for (int j = 0; j < 4; ++j) {
    const int n = n0 + 16 * j + (lane & 15);
    const float bj = bias[n];
#pragma unroll
    for (int r = 0; r < 8; ++r) {
      const int rowA = m0 + half * 8 + r;
      Out[(size_t)rowA * H_DIM + n]        = acc[0][j][r] + bj;
      Out[(size_t)(rowA + 16) * H_DIM + n] = acc[1][j][r] + bj;
    }
  }
}

// ---- Phase 2: one recurrent step ------------------------------------------
// Wave tile 64x16: 4 A-frags (LDS) share a single B-frag (global, L2-hot),
// so W_hh is fetched exactly once per WG. K is processed in 8 panels of 128,
// pipelined against the in-order async staging of h_prev into LDS.

__device__ __forceinline__ void rnn_block(const __bf16* ldsH,
                                          const __bf16* __restrict__ Whh,
                                          int n0, int k0base, v8f* acc) {
#pragma unroll
  for (int k0 = k0base; k0 < k0base + 128; k0 += 32) {
    v16bf b = load_b_bf16(Whh, H_DIM, n0, k0);
#pragma unroll
    for (int mi = 0; mi < 4; ++mi) {
      v16bf a = load_a_from_bf16(ldsH, H_DIM, mi * 16, k0);   // ds_load_b128
      acc[mi] = WMMA_BF16(a, b, acc[mi]);
    }
  }
}

// Wait for this wave's async chunks of panels <= P (in-order completion),
// then barrier so every wave's chunks of those panels are visible in LDS.
#define RNN_PANEL(P, WAITIMM)                                   \
  do {                                                          \
    asm volatile("s_wait_asynccnt " #WAITIMM ::: "memory");     \
    __syncthreads();                                            \
    rnn_block(ldsH, Whh, n0, (P) * 128, acc);                   \
  } while (0)

__global__ __launch_bounds__(256) void rnn_step(
    const __bf16* __restrict__ Hprev,  // [B, H] bf16
    const __bf16* __restrict__ Whh,    // [H, H] bf16
    float*        __restrict__ Xp,     // d_out + t*H ; [b][n] at b*(S*H)+n
    __bf16*       __restrict__ Hnext)  // [B, H] bf16
{
  extern __shared__ __bf16 ldsH[];     // B_DIM * H_DIM bf16 = 128 KB (dynamic)

  // ---- issue async staging in K-panel order: 8 panels x (4 chunks/thread).
  // Panel p = K columns [p*128, p*128+128) of all 64 rows = 16 KB.
  {
    const unsigned lbase = (unsigned)(uintptr_t)ldsH;  // low 32b = LDS offset
    const char*    gbase = (const char*)Hprev;
#pragma unroll
    for (int p = 0; p < 8; ++p) {
#pragma unroll
      for (int j = 0; j < 4; ++j) {
        const unsigned c    = threadIdx.x + (unsigned)j * 256u;   // 0..1023
        const unsigned row  = c >> 4;                             // 0..63
        const unsigned off  = row * (H_DIM * 2u) + (unsigned)p * 256u
                              + (c & 15u) * 16u;
        asm volatile("global_load_async_to_lds_b128 %0, %1, off"
                     :: "v"(lbase + off), "v"(gbase + off) : "memory");
      }
    }
  }

  const int waveId = threadIdx.x >> 5;
  const int n0 = blockIdx.x * 128 + waveId * 16;       // 8 waves x 16 cols

  v8f acc[4];
#pragma unroll
  for (int mi = 0; mi < 4; ++mi) acc[mi] = v8f{};

  RNN_PANEL(0, 28); RNN_PANEL(1, 24); RNN_PANEL(2, 20); RNN_PANEL(3, 16);
  RNN_PANEL(4, 12); RNN_PANEL(5, 8);  RNN_PANEL(6, 4);  RNN_PANEL(7, 0);

  const int lane = threadIdx.x & 31;
  const int half = lane >> 4;
  const int n    = n0 + (lane & 15);
  const size_t ldo = (size_t)S_DIM * H_DIM;            // xp row stride
#pragma unroll
  for (int mi = 0; mi < 4; ++mi) {
#pragma unroll
    for (int r = 0; r < 8; ++r) {
      const int row = mi * 16 + half * 8 + r;
      float v = fmaxf(acc[mi][r] + Xp[(size_t)row * ldo + n], 0.f);
      Xp[(size_t)row * ldo + n]    = v;
      Hnext[(size_t)row * H_DIM + n] = (__bf16)v;
    }
  }
}

// ---- small utility kernels -------------------------------------------------
__global__ void k_f32_to_bf16(const float* __restrict__ src,
                              __bf16* __restrict__ dst, int n) {
  int i = blockIdx.x * blockDim.x + threadIdx.x;
  if (i < n) dst[i] = (__bf16)src[i];
}

__global__ void k_bias_sum(const float* __restrict__ a,
                           const float* __restrict__ b,
                           float* __restrict__ dst, int n) {
  int i = blockIdx.x * blockDim.x + threadIdx.x;
  if (i < n) dst[i] = a[i] + b[i];
}

__global__ void k_copy_hn(const float* __restrict__ outs,
                          float* __restrict__ hn) {
  int i = blockIdx.x * blockDim.x + threadIdx.x;
  if (i < B_DIM * H_DIM) {
    int b = i / H_DIM, h = i - b * H_DIM;
    hn[i] = outs[(size_t)b * S_DIM * H_DIM + (size_t)(S_DIM - 1) * H_DIM + h];
  }
}

// ---------------------------------------------------------------------------
extern "C" void kernel_launch(void* const* d_in, const int* in_sizes, int n_in,
                              void* d_out, int out_size, void* d_ws, size_t ws_size,
                              hipStream_t stream) {
  (void)in_sizes; (void)n_in; (void)out_size; (void)ws_size;
  const float* x   = (const float*)d_in[0];   // [B,S,I]
  const float* h0  = (const float*)d_in[1];   // [1,B,H]
  const float* wih = (const float*)d_in[2];   // [H,I]
  const float* whh = (const float*)d_in[3];   // [H,H]
  const float* bih = (const float*)d_in[4];   // [H]
  const float* bhh = (const float*)d_in[5];   // [H]
  float* out = (float*)d_out;                 // [B,S,H] ++ [1,B,H]

  // Workspace layout (~4.3 MB)
  char* ws = (char*)d_ws;
  __bf16* wih_bf = (__bf16*)ws;                                          // 2 MB
  __bf16* whh_bf = (__bf16*)(ws + (size_t)H_DIM * I_DIM * 2);            // 2 MB
  float*  bias   = (float*) (ws + (size_t)2 * H_DIM * I_DIM * 2);        // 4 KB
  __bf16* hA     = (__bf16*)((char*)bias + H_DIM * sizeof(float));       // 128 KB
  __bf16* hB     = hA + (size_t)B_DIM * H_DIM;                           // 128 KB

  const int nW = H_DIM * I_DIM;
  k_f32_to_bf16<<<(nW + 255) / 256, 256, 0, stream>>>(wih, wih_bf, nW);
  k_f32_to_bf16<<<(nW + 255) / 256, 256, 0, stream>>>(whh, whh_bf, nW);
  k_f32_to_bf16<<<(B_DIM * H_DIM + 255) / 256, 256, 0, stream>>>(h0, hA, B_DIM * H_DIM);
  k_bias_sum<<<(H_DIM + 255) / 256, 256, 0, stream>>>(bih, bhh, bias, H_DIM);

  // Phase 1: x_proj into d_out. Grid: N/256 x M/64 workgroups of 8 waves.
  dim3 g1(H_DIM / 256, (B_DIM * S_DIM) / 64);
  xproj_gemm<<<g1, 256, 0, stream>>>(x, wih_bf, bias, out);

  // Phase 2: 512 dependent steps, ordered by the stream (graph-replayed).
  const size_t ldsBytes = (size_t)B_DIM * H_DIM * sizeof(__bf16);  // 128 KB
  for (int t = 0; t < S_DIM; ++t) {
    const __bf16* hp = (t & 1) ? hB : hA;
    __bf16*       hn = (t & 1) ? hA : hB;
    rnn_step<<<dim3(H_DIM / 128), 256, ldsBytes, stream>>>(
        hp, whh_bf, out + (size_t)t * H_DIM, hn);
  }

  // h_n = outputs[:, S-1, :]
  k_copy_hn<<<(B_DIM * H_DIM + 255) / 256, 256, 0, stream>>>(
      out, out + (size_t)B_DIM * S_DIM * H_DIM);
}
